// multi_branch_LadderBlock_69166153335118
// MI455X (gfx1250) — matrix operand order repaired and checked by reading, not compiled, not run on hardware
//
#include <hip/hip_runtime.h>
#include <hip/hip_bf16.h>

// ---------------------------------------------------------------------------
// Swin LadderBlock for MI455X (gfx1250), all GEMMs via v_wmma_f32_16x16x32_bf16
// ---------------------------------------------------------------------------

typedef __bf16 bf16;
typedef bf16 v16bf __attribute__((ext_vector_type(16)));
typedef float v8f  __attribute__((ext_vector_type(8)));

#define WMMA_BF16(a, b, c) \
  __builtin_amdgcn_wmma_f32_16x16x32_bf16(false, (a), false, (b), (short)0, (c), false, false)

union FragU { uint4 u4[2]; v16bf v; };

// A fragment (16x32 bf16), row-major source with leading dim ld.
// ISA layout: lane<16 holds K={0..7,16..23}, lane>=16 holds K={8..15,24..31}.
__device__ __forceinline__ v16bf load_fragA(const bf16* p, int ld, int row0, int k0, int lane) {
  int r = lane & 15, hl = lane >> 4;
  const bf16* q = p + (size_t)(row0 + r) * ld + k0 + hl * 8;
  FragU f;
  f.u4[0] = *(const uint4*)(q);
  f.u4[1] = *(const uint4*)(q + 16);
  return f.v;
}

// B fragment (32x16 bf16) loaded from (N x K) row-major source (lane = column).
// ISA layout: lane<16 holds K={0..7,16..23}, lane>=16 holds K={8..15,24..31}.
__device__ __forceinline__ v16bf load_fragB(const bf16* p, int ld, int col0, int k0, int lane) {
  int r = lane & 15, hl = lane >> 4;
  const bf16* q = p + (size_t)(col0 + r) * ld + k0 + hl * 8;
  FragU f;
  f.u4[0] = *(const uint4*)(q);
  f.u4[1] = *(const uint4*)(q + 16);
  return f.v;
}

// 32x32 wave tile: C[row0..+32, col0..+32] += A(MxK) * W(NxK)^T  (lda = ldb = K)
__device__ __forceinline__ void gemm_accum(const bf16* A, const bf16* W, int K,
                                           int row0, int col0, int lane, v8f acc[2][2]) {
  for (int k0 = 0; k0 < K; k0 += 32) {
    v16bf a0 = load_fragA(A, K, row0,      k0, lane);
    v16bf a1 = load_fragA(A, K, row0 + 16, k0, lane);
    v16bf b0 = load_fragB(W, K, col0,      k0, lane);
    v16bf b1 = load_fragB(W, K, col0 + 16, k0, lane);
    acc[0][0] = WMMA_BF16(a0, b0, acc[0][0]);
    acc[0][1] = WMMA_BF16(a0, b1, acc[0][1]);
    acc[1][0] = WMMA_BF16(a1, b0, acc[1][0]);
    acc[1][1] = WMMA_BF16(a1, b1, acc[1][1]);
  }
}

__device__ __forceinline__ float gelu_exact(float v) {
  return 0.5f * v * (1.0f + erff(v * 0.70710678118654752f));
}
__device__ __forceinline__ int sregion(int h) {  // thresholds: 0..48 / 49..52 / 53..55
  return h < 49 ? 0 : (h < 53 ? 1 : 2);
}

#define GEMM_PROLOGUE                                   \
  int tid = threadIdx.x; int wid = tid >> 5;            \
  int lane = tid & 31;                                  \
  int wm = wid >> 1, wn = wid & 1;                      \
  int row0 = blockIdx.x * 128 + wm * 32;                \
  int col0 = blockIdx.y * 64 + wn * 32;                 \
  int r = lane & 15, hl = lane >> 4;                    \
  v8f acc[2][2] = {};

// ---------------------------------------------------------------------------
// fp32 -> bf16 copy (weights)
// ---------------------------------------------------------------------------
__global__ void cvt_kernel(const float* __restrict__ s, bf16* __restrict__ d, int n) {
  int i = blockIdx.x * 256 + threadIdx.x;
  if (i < n) d[i] = (bf16)s[i];
}

// ---------------------------------------------------------------------------
// LN1 + cyclic shift(-3,-3) + window partition -> a0 bf16 (2048, 49, 384)
// one wave per token row; 8 rows per block
// ---------------------------------------------------------------------------
__global__ __launch_bounds__(256) void ln1_window_kernel(const float* __restrict__ x,
                                                         const float* __restrict__ g,
                                                         const float* __restrict__ bta,
                                                         bf16* __restrict__ a0) {
  int wid = threadIdx.x >> 5, lane = threadIdx.x & 31;
  int row = blockIdx.x * 8 + wid;               // < 100352
  const float* xr = x + (size_t)row * 384;
  float vals[12];
  float s = 0.f;
#pragma unroll
  for (int i = 0; i < 12; ++i) { vals[i] = xr[lane + 32 * i]; s += vals[i]; }
#pragma unroll
  for (int off = 16; off; off >>= 1) s += __shfl_xor(s, off, 32);
  float mean = s * (1.0f / 384.0f);
  float vs = 0.f;
#pragma unroll
  for (int i = 0; i < 12; ++i) { float d = vals[i] - mean; vs += d * d; }
#pragma unroll
  for (int off = 16; off; off >>= 1) vs += __shfl_xor(vs, off, 32);
  float rstd = rsqrtf(vs * (1.0f / 384.0f) + 1e-5f);

  int b = row / 3136, l = row % 3136, h = l / 56, w = l % 56;
  int hs = (h + 53) % 56, wsft = (w + 53) % 56;          // roll(-3)
  int win = b * 64 + (hs / 7) * 8 + (wsft / 7);
  int tok = (hs % 7) * 7 + (wsft % 7);
  bf16* dst = a0 + ((size_t)win * 49 + tok) * 384;
#pragma unroll
  for (int i = 0; i < 12; ++i) {
    int c = lane + 32 * i;
    dst[c] = (bf16)((vals[i] - mean) * rstd * g[c] + bta[c]);
  }
}

// ---------------------------------------------------------------------------
// QKV GEMM: a0(100352x384) @ qkv_w^T(384x1152) + bias
// epilogue scatters into gq/gk (win,head,tok,32) and gvt (win,head,32,tok49)
// ---------------------------------------------------------------------------
__global__ __launch_bounds__(256) void gemm_qkv_kernel(const bf16* __restrict__ A,
                                                       const bf16* __restrict__ W,
                                                       const float* __restrict__ bias,
                                                       bf16* __restrict__ gq,
                                                       bf16* __restrict__ gk,
                                                       bf16* __restrict__ gvt) {
  GEMM_PROLOGUE
  gemm_accum(A, W, 384, row0, col0, lane, acc);
#pragma unroll
  for (int mi = 0; mi < 2; ++mi)
#pragma unroll
    for (int ni = 0; ni < 2; ++ni) {
      int n = col0 + ni * 16 + r;
      float bv = bias[n];
      int which = n / 384, head = (n % 384) >> 5, d = n & 31;
#pragma unroll
      for (int e = 0; e < 8; ++e) {
        int m = row0 + mi * 16 + hl * 8 + e;
        int win = m / 49, tok = m % 49;
        float v = acc[mi][ni][e] + bv;
        if (which == 0)
          gq[(((size_t)win * 12 + head) * 49 + tok) * 32 + d] = (bf16)(v * 0.17677669529663687f);
        else if (which == 1)
          gk[(((size_t)win * 12 + head) * 49 + tok) * 32 + d] = (bf16)v;
        else
          gvt[(((size_t)win * 12 + head) * 32 + d) * 49 + tok] = (bf16)v;
      }
    }
}

// ---------------------------------------------------------------------------
// Attention: one block (4 waves) per (window, head).
// S = Q K^T (+rpb, +shift mask), softmax, O = P V. All matmuls via WMMA from LDS.
// ---------------------------------------------------------------------------
__global__ __launch_bounds__(128) void attn_kernel(const bf16* __restrict__ gq,
                                                   const bf16* __restrict__ gk,
                                                   const bf16* __restrict__ gvt,
                                                   const float* __restrict__ rpb_table,
                                                   bf16* __restrict__ attnout) {
  __shared__ __attribute__((aligned(16))) bf16 sQ[64 * 32];
  __shared__ __attribute__((aligned(16))) bf16 sK[64 * 32];
  __shared__ __attribute__((aligned(16))) bf16 sVt[32 * 64];
  __shared__ __attribute__((aligned(16))) float sS[64][64];
  __shared__ __attribute__((aligned(16))) bf16 sP[64][64];

  int tid = threadIdx.x;
  int win = blockIdx.x / 12, head = blockIdx.x % 12;
  const bf16* qb = gq + ((size_t)win * 12 + head) * 49 * 32;
  const bf16* kb = gk + ((size_t)win * 12 + head) * 49 * 32;
  const bf16* vb = gvt + ((size_t)win * 12 + head) * 32 * 49;

  for (int i = tid; i < 64 * 32; i += 128) {
    int t = i >> 5, d = i & 31;
    bf16 z = (bf16)0.0f;
    sQ[i] = (t < 49) ? qb[t * 32 + d] : z;
    sK[i] = (t < 49) ? kb[t * 32 + d] : z;
  }
  for (int i = tid; i < 32 * 64; i += 128) {
    int d = i >> 6, t = i & 63;
    sVt[i] = (t < 49) ? vb[d * 49 + t] : (bf16)0.0f;
  }
  __syncthreads();

  int wv = tid >> 5, lane = tid & 31;
  int r = lane & 15, hl = lane >> 4;
  int row0 = wv * 16;

  // S = Q K^T : wave computes rows [row0, row0+16), all 64 columns
  v16bf aq = load_fragA(sQ, 32, row0, 0, lane);
  v8f sacc[4];
#pragma unroll
  for (int nt = 0; nt < 4; ++nt) {
    v16bf bk = load_fragB(sK, 32, nt * 16, 0, lane);
    v8f z = {};
    sacc[nt] = WMMA_BF16(aq, bk, z);
  }

  int iw = win & 63, wh = iw >> 3, ww = iw & 7;
#pragma unroll
  for (int nt = 0; nt < 4; ++nt)
#pragma unroll
    for (int e = 0; e < 8; ++e) {
      int i = row0 + hl * 8 + e;
      int j = nt * 16 + r;
      float v = sacc[nt][e];
      if (i < 49 && j < 49) {
        int ri = i / 7, ci = i % 7, rj = j / 7, cj = j % 7;
        int idx = (ri - rj + 6) * 13 + (ci - cj + 6);
        v += rpb_table[idx * 12 + head];
        int regi = 3 * sregion(wh * 7 + ri) + sregion(ww * 7 + ci);
        int regj = 3 * sregion(wh * 7 + rj) + sregion(ww * 7 + cj);
        if (regi != regj) v -= 100.0f;
      }
      sS[i][j] = v;
    }
  __syncthreads();

  // softmax over 49 real columns, one thread per row
  if (tid < 64) {
    int row = tid;
    if (row < 49) {
      float mx = -1e30f;
      for (int j = 0; j < 49; ++j) mx = fmaxf(mx, sS[row][j]);
      float sum = 0.f;
      for (int j = 0; j < 49; ++j) { float e = __expf(sS[row][j] - mx); sS[row][j] = e; sum += e; }
      float inv = 1.0f / sum;
      for (int j = 0; j < 64; ++j) sP[row][j] = (bf16)(j < 49 ? sS[row][j] * inv : 0.0f);
    } else {
      for (int j = 0; j < 64; ++j) sP[row][j] = (bf16)0.0f;
    }
  }
  __syncthreads();

  // O = P V : rows [row0,row0+16) x HD=32, K=64 (2 steps)
  v8f oacc[2] = {};
  for (int k0 = 0; k0 < 64; k0 += 32) {
    v16bf ap = load_fragA(&sP[0][0], 64, row0, k0, lane);
#pragma unroll
    for (int ni = 0; ni < 2; ++ni) {
      v16bf bv = load_fragB(sVt, 64, ni * 16, k0, lane);
      oacc[ni] = WMMA_BF16(ap, bv, oacc[ni]);
    }
  }
  bf16* ob = attnout + (size_t)win * 49 * 384 + head * 32;
#pragma unroll
  for (int ni = 0; ni < 2; ++ni)
#pragma unroll
    for (int e = 0; e < 8; ++e) {
      int i = row0 + hl * 8 + e;
      if (i < 49) ob[(size_t)i * 384 + ni * 16 + r] = (bf16)oacc[ni][e];
    }
}

// ---------------------------------------------------------------------------
// proj GEMM + window-reverse + roll(+3,+3) + residual -> x1 (f32) and x1b (bf16)
// ---------------------------------------------------------------------------
__global__ __launch_bounds__(256) void gemm_proj_kernel(const bf16* __restrict__ A,
                                                        const bf16* __restrict__ W,
                                                        const float* __restrict__ bias,
                                                        const float* __restrict__ xin,
                                                        float* __restrict__ x1,
                                                        bf16* __restrict__ x1b) {
  GEMM_PROLOGUE
  gemm_accum(A, W, 384, row0, col0, lane, acc);
#pragma unroll
  for (int mi = 0; mi < 2; ++mi)
#pragma unroll
    for (int ni = 0; ni < 2; ++ni) {
      int n = col0 + ni * 16 + r;
      float bv = bias[n];
#pragma unroll
      for (int e = 0; e < 8; ++e) {
        int m = row0 + mi * 16 + hl * 8 + e;
        int win = m / 49, tok = m % 49;
        int b = win >> 6, iw = win & 63, wh = iw >> 3, wwp = iw & 7;
        int rr = tok / 7, cc = tok % 7;
        int h = (wh * 7 + rr + 3) % 56;
        int w = (wwp * 7 + cc + 3) % 56;
        size_t a = ((size_t)b * 3136 + h * 56 + w) * 384 + n;
        float v = acc[mi][ni][e] + bv + xin[a];
        x1[a] = v;
        x1b[a] = (bf16)v;
      }
    }
}

// ---------------------------------------------------------------------------
// SE GEMM 1: relu(x1b @ se1^T + b) -> g1 bf16 (M x 192)
// ---------------------------------------------------------------------------
__global__ __launch_bounds__(256) void gemm_se1_kernel(const bf16* __restrict__ A,
                                                       const bf16* __restrict__ W,
                                                       const float* __restrict__ bias,
                                                       bf16* __restrict__ g1) {
  GEMM_PROLOGUE
  gemm_accum(A, W, 384, row0, col0, lane, acc);
#pragma unroll
  for (int mi = 0; mi < 2; ++mi)
#pragma unroll
    for (int ni = 0; ni < 2; ++ni) {
      int n = col0 + ni * 16 + r;
      float bv = bias[n];
#pragma unroll
      for (int e = 0; e < 8; ++e) {
        int m = row0 + mi * 16 + hl * 8 + e;
        float v = acc[mi][ni][e] + bv;
        g1[(size_t)m * 192 + n] = (bf16)(v > 0.f ? v : 0.f);
      }
    }
}

// ---------------------------------------------------------------------------
// SE GEMM 2: x2 = x1 * sigmoid(g1 @ se2^T + b)  (K = 192)
// ---------------------------------------------------------------------------
__global__ __launch_bounds__(256) void gemm_se2_kernel(const bf16* __restrict__ A,
                                                       const bf16* __restrict__ W,
                                                       const float* __restrict__ bias,
                                                       const float* __restrict__ x1,
                                                       float* __restrict__ x2) {
  GEMM_PROLOGUE
  gemm_accum(A, W, 192, row0, col0, lane, acc);
#pragma unroll
  for (int mi = 0; mi < 2; ++mi)
#pragma unroll
    for (int ni = 0; ni < 2; ++ni) {
      int n = col0 + ni * 16 + r;
      float bv = bias[n];
#pragma unroll
      for (int e = 0; e < 8; ++e) {
        int m = row0 + mi * 16 + hl * 8 + e;
        float v = acc[mi][ni][e] + bv;
        float g = 1.0f / (1.0f + __expf(-v));
        size_t a = (size_t)m * 384 + n;
        x2[a] = x1[a] * g;
      }
    }
}

// ---------------------------------------------------------------------------
// LN2 -> y bf16 (linear layout)
// ---------------------------------------------------------------------------
__global__ __launch_bounds__(256) void ln2_kernel(const float* __restrict__ x,
                                                  const float* __restrict__ g,
                                                  const float* __restrict__ bta,
                                                  bf16* __restrict__ y) {
  int wid = threadIdx.x >> 5, lane = threadIdx.x & 31;
  int row = blockIdx.x * 8 + wid;
  const float* xr = x + (size_t)row * 384;
  float vals[12];
  float s = 0.f;
#pragma unroll
  for (int i = 0; i < 12; ++i) { vals[i] = xr[lane + 32 * i]; s += vals[i]; }
#pragma unroll
  for (int off = 16; off; off >>= 1) s += __shfl_xor(s, off, 32);
  float mean = s * (1.0f / 384.0f);
  float vs = 0.f;
#pragma unroll
  for (int i = 0; i < 12; ++i) { float d = vals[i] - mean; vs += d * d; }
#pragma unroll
  for (int off = 16; off; off >>= 1) vs += __shfl_xor(vs, off, 32);
  float rstd = rsqrtf(vs * (1.0f / 384.0f) + 1e-5f);
  bf16* dst = y + (size_t)row * 384;
#pragma unroll
  for (int i = 0; i < 12; ++i) {
    int c = lane + 32 * i;
    dst[c] = (bf16)((vals[i] - mean) * rstd * g[c] + bta[c]);
  }
}

// ---------------------------------------------------------------------------
// c1 GEMM: z = gelu(y @ c1^T + b) * bn1_g + bn1_b  (f32, NHWC)
// ---------------------------------------------------------------------------
__global__ __launch_bounds__(256) void gemm_c1_kernel(const bf16* __restrict__ A,
                                                      const bf16* __restrict__ W,
                                                      const float* __restrict__ bias,
                                                      const float* __restrict__ gam,
                                                      const float* __restrict__ bet,
                                                      float* __restrict__ z) {
  GEMM_PROLOGUE
  gemm_accum(A, W, 384, row0, col0, lane, acc);
#pragma unroll
  for (int mi = 0; mi < 2; ++mi)
#pragma unroll
    for (int ni = 0; ni < 2; ++ni) {
      int n = col0 + ni * 16 + r;
      float bv = bias[n], gg = gam[n], bb = bet[n];
#pragma unroll
      for (int e = 0; e < 8; ++e) {
        int m = row0 + mi * 16 + hl * 8 + e;
        float v = acc[mi][ni][e] + bv;
        z[(size_t)m * 384 + n] = gelu_exact(v) * gg + bb;
      }
    }
}

// ---------------------------------------------------------------------------
// depthwise 3x3 SAME (NHWC) + gelu + bn2 affine -> t2 bf16
// ---------------------------------------------------------------------------
__global__ __launch_bounds__(256) void dwconv_kernel(const float* __restrict__ z,
                                                     const float* __restrict__ wt,
                                                     const float* __restrict__ bias,
                                                     const float* __restrict__ gam,
                                                     const float* __restrict__ bet,
                                                     bf16* __restrict__ t2) {
  size_t idx = (size_t)blockIdx.x * 256 + threadIdx.x;   // < 38535168
  int ch = (int)(idx % 384);
  size_t rest = idx / 384;
  int w = (int)(rest % 56);
  int h = (int)((rest / 56) % 56);
  int b = (int)(rest / 3136);
  float s = 0.f;
#pragma unroll
  for (int kh = 0; kh < 3; ++kh) {
    int hh = h + kh - 1;
    if ((unsigned)hh >= 56u) continue;
#pragma unroll
    for (int kw = 0; kw < 3; ++kw) {
      int wwp = w + kw - 1;
      if ((unsigned)wwp >= 56u) continue;
      s += z[((size_t)b * 3136 + hh * 56 + wwp) * 384 + ch] * wt[ch * 9 + kh * 3 + kw];
    }
  }
  float v = s + bias[ch];
  t2[idx] = (bf16)(gelu_exact(v) * gam[ch] + bet[ch]);
}

// ---------------------------------------------------------------------------
// c2 GEMM: out = x2 + (t2 @ c2^T + b) * bn3_g + bn3_b
// ---------------------------------------------------------------------------
__global__ __launch_bounds__(256) void gemm_c2_kernel(const bf16* __restrict__ A,
                                                      const bf16* __restrict__ W,
                                                      const float* __restrict__ bias,
                                                      const float* __restrict__ gam,
                                                      const float* __restrict__ bet,
                                                      const float* __restrict__ x2,
                                                      float* __restrict__ out) {
  GEMM_PROLOGUE
  gemm_accum(A, W, 384, row0, col0, lane, acc);
#pragma unroll
  for (int mi = 0; mi < 2; ++mi)
#pragma unroll
    for (int ni = 0; ni < 2; ++ni) {
      int n = col0 + ni * 16 + r;
      float bv = bias[n], gg = gam[n], bb = bet[n];
#pragma unroll
      for (int e = 0; e < 8; ++e) {
        int m = row0 + mi * 16 + hl * 8 + e;
        size_t a = (size_t)m * 384 + n;
        float v = acc[mi][ni][e] + bv;
        out[a] = x2[a] + v * gg + bb;
      }
    }
}

// ---------------------------------------------------------------------------
// host launch
// ---------------------------------------------------------------------------
extern "C" void kernel_launch(void* const* d_in, const int* in_sizes, int n_in,
                              void* d_out, int out_size, void* d_ws, size_t ws_size,
                              hipStream_t stream) {
  (void)in_sizes; (void)n_in; (void)out_size; (void)ws_size;
  const float* x      = (const float*)d_in[0];
  const float* n1g    = (const float*)d_in[1];
  const float* n1b    = (const float*)d_in[2];
  const float* qkv_w  = (const float*)d_in[3];
  const float* qkv_b  = (const float*)d_in[4];
  const float* rpb    = (const float*)d_in[5];
  const float* proj_w = (const float*)d_in[6];
  const float* proj_b = (const float*)d_in[7];
  const float* se1_w  = (const float*)d_in[8];
  const float* se1_b  = (const float*)d_in[9];
  const float* se2_w  = (const float*)d_in[10];
  const float* se2_b  = (const float*)d_in[11];
  const float* n2g    = (const float*)d_in[12];
  const float* n2b    = (const float*)d_in[13];
  const float* c1_w   = (const float*)d_in[14];
  const float* c1_b   = (const float*)d_in[15];
  const float* bn1g   = (const float*)d_in[16];
  const float* bn1b   = (const float*)d_in[17];
  const float* dw_w   = (const float*)d_in[18];
  const float* dw_b   = (const float*)d_in[19];
  const float* bn2g   = (const float*)d_in[20];
  const float* bn2b   = (const float*)d_in[21];
  const float* c2_w   = (const float*)d_in[22];
  const float* c2_b   = (const float*)d_in[23];
  const float* bn3g   = (const float*)d_in[24];
  const float* bn3b   = (const float*)d_in[25];
  float* out = (float*)d_out;

  const size_t M = 100352;  // 32 * 56 * 56
  char* ws = (char*)d_ws;
  size_t off = 0;
  auto alloc = [&](size_t bytes) { size_t o = off; off += (bytes + 255) & ~(size_t)255; return o; };

  bf16* wqkvb  = (bf16*)(ws + alloc((size_t)442368 * 2));
  bf16* wprojb = (bf16*)(ws + alloc((size_t)147456 * 2));
  bf16* wse1b  = (bf16*)(ws + alloc((size_t)73728 * 2));
  bf16* wse2b  = (bf16*)(ws + alloc((size_t)73728 * 2));
  bf16* wc1b   = (bf16*)(ws + alloc((size_t)147456 * 2));
  bf16* wc2b   = (bf16*)(ws + alloc((size_t)147456 * 2));
  bf16* Ra0  = (bf16*)(ws + alloc(M * 384 * 2));   // a0 -> attnout -> y
  bf16* Rgq  = (bf16*)(ws + alloc(M * 384 * 2));   // gq -> x1b -> t2
  bf16* Rgk  = (bf16*)(ws + alloc(M * 384 * 2));   // gk -> g1
  bf16* Rgvt = (bf16*)(ws + alloc(M * 384 * 2));   // vT
  float* Rx1 = (float*)(ws + alloc(M * 384 * 4));  // x1 -> z
  float* Rx2 = (float*)(ws + alloc(M * 384 * 4));  // x2

  // weights to bf16 (N x K row-major == WMMA B-fragment friendly)
  cvt_kernel<<<(442368 + 255) / 256, 256, 0, stream>>>(qkv_w, wqkvb, 442368);
  cvt_kernel<<<(147456 + 255) / 256, 256, 0, stream>>>(proj_w, wprojb, 147456);
  cvt_kernel<<<(73728 + 255) / 256, 256, 0, stream>>>(se1_w, wse1b, 73728);
  cvt_kernel<<<(73728 + 255) / 256, 256, 0, stream>>>(se2_w, wse2b, 73728);
  cvt_kernel<<<(147456 + 255) / 256, 256, 0, stream>>>(c1_w, wc1b, 147456);
  cvt_kernel<<<(147456 + 255) / 256, 256, 0, stream>>>(c2_w, wc2b, 147456);

  ln1_window_kernel<<<12544, 256, 0, stream>>>(x, n1g, n1b, Ra0);
  gemm_qkv_kernel<<<dim3(784, 18), 256, 0, stream>>>(Ra0, wqkvb, qkv_b, Rgq, Rgk, Rgvt);
  attn_kernel<<<24576, 128, 0, stream>>>(Rgq, Rgk, Rgvt, rpb, Ra0);
  gemm_proj_kernel<<<dim3(784, 6), 256, 0, stream>>>(Ra0, wprojb, proj_b, x, Rx1, Rgq);
  gemm_se1_kernel<<<dim3(784, 3), 256, 0, stream>>>(Rgq, wse1b, se1_b, Rgk);
  gemm_se2_kernel<<<dim3(784, 6), 256, 0, stream>>>(Rgk, wse2b, se2_b, Rx1, Rx2);
  ln2_kernel<<<12544, 256, 0, stream>>>(Rx2, n2g, n2b, Ra0);
  gemm_c1_kernel<<<dim3(784, 6), 256, 0, stream>>>(Ra0, wc1b, c1_b, bn1g, bn1b, Rx1);
  dwconv_kernel<<<150528, 256, 0, stream>>>(Rx1, dw_w, dw_b, bn2g, bn2b, Rgq);
  gemm_c2_kernel<<<dim3(784, 6), 256, 0, stream>>>(Rgq, wc2b, c2_b, bn3g, bn3b, Rx2, out);
}
